// MaskedCrossAttention_46127948759275
// MI455X (gfx1250) — matrix operand order repaired and checked by reading, not compiled, not run on hardware
//
#include <hip/hip_runtime.h>
#include <hip/hip_bf16.h>

// MaskedCrossAttention for MI455X (gfx1250): bf16 WMMA everywhere,
// frame-local masked attention (each query attends to exactly one 256-key frame).
#define B_  2
#define N_  2048
#define T_  8
#define M_  256
#define D_  1024
#define H_  16
#define HD_ 64
#define S_  (T_*M_)          // 2048 keys total

typedef __attribute__((ext_vector_type(16))) __bf16 v16bf;
typedef __attribute__((ext_vector_type(8)))  __bf16 v8bf;
typedef __attribute__((ext_vector_type(4)))  __bf16 v4bf;
typedef __attribute__((ext_vector_type(8)))  float  v8f;

#define OUT_BF16 0
#define OUT_VT   1
#define OUT_F32  2

static __device__ __forceinline__ v16bf cat16(v8bf lo, v8bf hi) {
  v16bf r;
#pragma unroll
  for (int j = 0; j < 8; ++j) { r[j] = lo[j]; r[8 + j] = hi[j]; }
  return r;
}
static __device__ __forceinline__ v8f zero8() {
  v8f z;
#pragma unroll
  for (int i = 0; i < 8; ++i) z[i] = 0.f;
  return z;
}
static __device__ __forceinline__ v8f wmma_bf16(v16bf a, v16bf b, v8f c) {
  return __builtin_amdgcn_wmma_f32_16x16x32_bf16(false, a, false, b, (short)0, c,
                                                 false, false);
}

// ---------------- LayerNorm + cast to bf16 (one row per block) ----------------
__global__ __launch_bounds__(256)
void ln_cast_kernel(const float* __restrict__ x, const float* __restrict__ gamma,
                    const float* __restrict__ beta, __bf16* __restrict__ y) {
  __shared__ float red[256];
  const int row = blockIdx.x, tid = threadIdx.x;
  const float* xr = x + (size_t)row * D_;
  float4 v = *(const float4*)(xr + tid * 4);
  red[tid] = v.x + v.y + v.z + v.w;
  __syncthreads();
  for (int st = 128; st > 0; st >>= 1) {
    if (tid < st) red[tid] += red[tid + st];
    __syncthreads();
  }
  float mu = red[0] * (1.f / D_);
  __syncthreads();
  float d0 = v.x - mu, d1 = v.y - mu, d2 = v.z - mu, d3 = v.w - mu;
  red[tid] = d0 * d0 + d1 * d1 + d2 * d2 + d3 * d3;
  __syncthreads();
  for (int st = 128; st > 0; st >>= 1) {
    if (tid < st) red[tid] += red[tid + st];
    __syncthreads();
  }
  float rstd = rsqrtf(red[0] * (1.f / D_) + 1e-5f);
  const int c = tid * 4;
  v4bf p;
  p[0] = (__bf16)(d0 * rstd * gamma[c + 0] + beta[c + 0]);
  p[1] = (__bf16)(d1 * rstd * gamma[c + 1] + beta[c + 1]);
  p[2] = (__bf16)(d2 * rstd * gamma[c + 2] + beta[c + 2]);
  p[3] = (__bf16)(d3 * rstd * gamma[c + 3] + beta[c + 3]);
  *(v4bf*)(y + (size_t)row * D_ + c) = p;
}

// ---------------- elementwise fp32 -> bf16 cast (media) ----------------
__global__ __launch_bounds__(256)
void cast_bf16_kernel(const float* __restrict__ x, __bf16* __restrict__ y) {
  const int i = (blockIdx.x * 256 + threadIdx.x) * 4;
  float4 v = *(const float4*)(x + i);
  v4bf p;
  p[0] = (__bf16)v.x; p[1] = (__bf16)v.y; p[2] = (__bf16)v.z; p[3] = (__bf16)v.w;
  *(v4bf*)(y + i) = p;
}

// ---------------- cumsum of media_locations -> frame index per token ----------------
__global__ void frames_kernel(const unsigned char* __restrict__ loc,
                              int* __restrict__ frame) {
  if (threadIdx.x == 0) {
    const unsigned char* L = loc + (size_t)blockIdx.x * N_;
    int* F = frame + (size_t)blockIdx.x * N_;
    int run = 0;
    for (int n = 0; n < N_; ++n) { run += (L[n] != 0); F[n] = run - 1; }
  }
}

// ---------------- generic bf16 WMMA GEMM: C[Mtot,1024] = A @ W + bias ----------------
// 8 waves/WG, WG tile = 128 rows x 64 cols, W tile staged transposed in LDS.
__global__ __launch_bounds__(256)
void gemm_wmma_kernel(const __bf16* __restrict__ A, const float* __restrict__ W,
                      const float* __restrict__ bias, void* __restrict__ Cout,
                      int mode) {
  __shared__ __bf16 lds_wt[64][40];           // [col][k], pitch 40 halfs (16B aligned)
  const int tid = threadIdx.x;
  const int lane = tid & 31, wave = tid >> 5;
  const int r = lane & 15, hi = lane >> 4;
  const int rowBase = blockIdx.x * 128 + wave * 16;
  const int n0 = blockIdx.y * 64;

  v8f acc[4];
#pragma unroll
  for (int c = 0; c < 4; ++c) acc[c] = zero8();

  for (int k0 = 0; k0 < D_; k0 += 32) {
    // stage W[k0:k0+32, n0:n0+64] transposed into LDS as bf16 (coalesced fp32 reads)
    const int kl = tid >> 3;                  // 0..31
    const int cb = (tid & 7) * 8;             // 0..56
    const float* wp = W + (size_t)(k0 + kl) * D_ + n0 + cb;
    float4 w0 = *(const float4*)wp;
    float4 w1 = *(const float4*)(wp + 4);
    lds_wt[cb + 0][kl] = (__bf16)w0.x; lds_wt[cb + 1][kl] = (__bf16)w0.y;
    lds_wt[cb + 2][kl] = (__bf16)w0.z; lds_wt[cb + 3][kl] = (__bf16)w0.w;
    lds_wt[cb + 4][kl] = (__bf16)w1.x; lds_wt[cb + 5][kl] = (__bf16)w1.y;
    lds_wt[cb + 6][kl] = (__bf16)w1.z; lds_wt[cb + 7][kl] = (__bf16)w1.w;
    __syncthreads();

    // A fragment (16x32), per ISA layout: two contiguous 8-half runs per lane
    const __bf16* ap = A + (size_t)(rowBase + r) * D_ + k0 + (hi ? 8 : 0);
    v16bf a = cat16(*(const v8bf*)ap, *(const v8bf*)(ap + 16));
#pragma unroll
    for (int c = 0; c < 4; ++c) {
      const __bf16* bp = &lds_wt[c * 16 + r][hi ? 8 : 0];
      v16bf b = cat16(*(const v8bf*)bp, *(const v8bf*)(bp + 16));
      acc[c] = wmma_bf16(a, b, acc[c]);
    }
    __syncthreads();
  }

  if (mode == OUT_VT) {
    // write V transposed: vt[((b*H + h)*HD + hd)*S + s], 8 contiguous s per lane
    const int row0 = rowBase + 8 * hi;
    const int bb = row0 / S_, s0 = row0 % S_;
#pragma unroll
    for (int c = 0; c < 4; ++c) {
      const int col = n0 + c * 16 + r;
      const float bv = bias[col];
      const int h = col >> 6, hd = col & 63;
      v8bf pk;
#pragma unroll
      for (int g = 0; g < 8; ++g) pk[g] = (__bf16)(acc[c][g] + bv);
      __bf16* vp = (__bf16*)Cout + ((size_t)(bb * H_ + h) * HD_ + hd) * S_ + s0;
      *(v8bf*)vp = pk;
    }
  } else {
#pragma unroll
    for (int c = 0; c < 4; ++c) {
      const int col = n0 + c * 16 + r;
      const float bv = bias[col];
#pragma unroll
      for (int g = 0; g < 8; ++g) {
        const size_t idx = (size_t)(rowBase + 8 * hi + g) * D_ + col;
        const float val = acc[c][g] + bv;
        if (mode == OUT_F32) ((float*)Cout)[idx] = val;
        else                 ((__bf16*)Cout)[idx] = (__bf16)val;
      }
    }
  }
}

// ---------------- masked attention: one wave per (b, h, 16-query tile) ----------------
// Each query row attends only to keys of its frame f = cumsum(media_locations)-1,
// i.e. keys [f*M, (f+1)*M). Online softmax across the (usually 1) frames in a tile.
__global__ __launch_bounds__(32)
void attn_kernel(const __bf16* __restrict__ q, const __bf16* __restrict__ k,
                 const __bf16* __restrict__ vt, const int* __restrict__ frame,
                 __bf16* __restrict__ o) {
  __shared__ __bf16 P[16][272];               // probs tile, pitch 272 halfs (16B aligned)
  const int lane = threadIdx.x & 31;
  const int r = lane & 15, hi = lane >> 4;
  const int b = blockIdx.z, h = blockIdx.y, q0 = blockIdx.x * 16;

  // Q A-fragments for the two hd chunks (hd 0..31 and 32..63)
  const __bf16* qrow = q + (size_t)(b * N_ + q0 + r) * D_ + h * HD_ + (hi ? 8 : 0);
  const v16bf aq0 = cat16(*(const v8bf*)qrow,        *(const v8bf*)(qrow + 16));
  const v16bf aq1 = cat16(*(const v8bf*)(qrow + 32), *(const v8bf*)(qrow + 48));

  // frames of the 8 rows this lane owns in the C/D layout (rows 8*hi + g)
  const int* fb = frame + (size_t)b * N_ + q0;
  int fr[8];
#pragma unroll
  for (int g = 0; g < 8; ++g) fr[g] = fb[8 * hi + g];
  int fmn = fr[0], fmx = fr[0];
#pragma unroll
  for (int g = 1; g < 8; ++g) { fmn = min(fmn, fr[g]); fmx = max(fmx, fr[g]); }
#pragma unroll
  for (int m = 16; m > 0; m >>= 1) {
    fmn = min(fmn, __shfl_xor(fmn, m, 32));
    fmx = max(fmx, __shfl_xor(fmx, m, 32));
  }
  if (fmn < 0) fmn = 0;

  v8f oacc[4];
#pragma unroll
  for (int c = 0; c < 4; ++c) oacc[c] = zero8();
  float rsum[8];
#pragma unroll
  for (int g = 0; g < 8; ++g) rsum[g] = 0.f;
  const float scale = 0.125f;                 // 1/sqrt(64)

  for (int f = fmn; f <= fmx; ++f) {
    const int s0 = f * M_;
    // scores S[16 x 256] for this frame: 16 key tiles x 2 hd-chunk WMMAs
    v8f sfr[16];
#pragma unroll
    for (int t = 0; t < 16; ++t) {
      const __bf16* kp =
          k + (size_t)(b * S_ + s0 + t * 16 + r) * D_ + h * HD_ + (hi ? 8 : 0);
      v16bf bk0 = cat16(*(const v8bf*)kp,        *(const v8bf*)(kp + 16));
      v16bf bk1 = cat16(*(const v8bf*)(kp + 32), *(const v8bf*)(kp + 48));
      v8f s = zero8();
      s = wmma_bf16(aq0, bk0, s);
      s = wmma_bf16(aq1, bk1, s);
      sfr[t] = s;
    }
    // row max (scaled) across the 256 keys of this frame
    float rmax[8];
#pragma unroll
    for (int g = 0; g < 8; ++g) {
      float mx = -1e30f;
#pragma unroll
      for (int t = 0; t < 16; ++t) mx = fmaxf(mx, sfr[t][g]);
      rmax[g] = mx * scale;
    }
#pragma unroll
    for (int m = 1; m < 16; m <<= 1)
#pragma unroll
      for (int g = 0; g < 8; ++g)
        rmax[g] = fmaxf(rmax[g], __shfl_xor(rmax[g], m, 32));

    // exp + mask rows whose frame != f; stage probs (bf16) in LDS; row sums
    float rs[8];
#pragma unroll
    for (int g = 0; g < 8; ++g) rs[g] = 0.f;
#pragma unroll
    for (int t = 0; t < 16; ++t)
#pragma unroll
      for (int g = 0; g < 8; ++g) {
        const float p = (fr[g] == f) ? __expf(sfr[t][g] * scale - rmax[g]) : 0.f;
        P[8 * hi + g][t * 16 + r] = (__bf16)p;
        rs[g] += p;
      }
#pragma unroll
    for (int m = 1; m < 16; m <<= 1)
#pragma unroll
      for (int g = 0; g < 8; ++g) rs[g] += __shfl_xor(rs[g], m, 32);
#pragma unroll
    for (int g = 0; g < 8; ++g) rsum[g] += rs[g];

    asm volatile("s_wait_dscnt 0" ::: "memory");  // LDS probs visible to own wave

    // O += P[16x256] @ V[256x64] : 8 key-chunks x 4 hd tiles
#pragma unroll
    for (int kk = 0; kk < 8; ++kk) {
      const __bf16* pp = &P[r][kk * 32 + (hi ? 8 : 0)];
      v16bf apf = cat16(*(const v8bf*)pp, *(const v8bf*)(pp + 16));
#pragma unroll
      for (int c = 0; c < 4; ++c) {
        const __bf16* vp = vt + ((size_t)(b * H_ + h) * HD_ + c * 16 + r) * S_ +
                           s0 + kk * 32 + (hi ? 8 : 0);
        v16bf bvf = cat16(*(const v8bf*)vp, *(const v8bf*)(vp + 16));
        oacc[c] = wmma_bf16(apf, bvf, oacc[c]);
      }
    }
  }

  // normalize and store O (row-major [b][n][h*64+hd], bf16)
  float inv[8];
#pragma unroll
  for (int g = 0; g < 8; ++g) inv[g] = (rsum[g] > 0.f) ? 1.f / rsum[g] : 0.f;
#pragma unroll
  for (int c = 0; c < 4; ++c)
#pragma unroll
    for (int g = 0; g < 8; ++g) {
      const size_t idx =
          (size_t)(b * N_ + q0 + 8 * hi + g) * D_ + h * HD_ + c * 16 + r;
      o[idx] = (__bf16)(oacc[c][g] * inv[g]);
    }
}

extern "C" void kernel_launch(void* const* d_in, const int* in_sizes, int n_in,
                              void* d_out, int out_size, void* d_ws, size_t ws_size,
                              hipStream_t stream) {
  (void)in_sizes; (void)n_in; (void)out_size; (void)ws_size;
  const float* text          = (const float*)d_in[0];
  const float* media         = (const float*)d_in[1];
  const unsigned char* loc   = (const unsigned char*)d_in[2];  // bool [B,N]
  const float* gamma         = (const float*)d_in[3];
  const float* beta          = (const float*)d_in[4];
  const float* wq            = (const float*)d_in[5];
  const float* wk            = (const float*)d_in[6];
  const float* wv            = (const float*)d_in[7];
  const float* bq            = (const float*)d_in[8];
  const float* bk            = (const float*)d_in[9];
  const float* bv            = (const float*)d_in[10];
  const float* wout          = (const float*)d_in[11];
  const float* bout          = (const float*)d_in[12];
  float* out                 = (float*)d_out;

  char* ws = (char*)d_ws;
  const size_t slab = (size_t)B_ * N_ * D_ * sizeof(__bf16);  // 8 MiB (B*N == B*S)
  __bf16* xbf  = (__bf16*)(ws + 0 * slab);
  __bf16* kvbf = (__bf16*)(ws + 1 * slab);
  __bf16* qbf  = (__bf16*)(ws + 2 * slab);
  __bf16* kbf  = (__bf16*)(ws + 3 * slab);
  __bf16* vtbf = (__bf16*)(ws + 4 * slab);
  int*    frame = (int*)(ws + 5 * slab);
  __bf16* obf  = xbf;  // reuse: x not needed after the Q projection

  ln_cast_kernel<<<dim3(B_ * N_), 256, 0, stream>>>(text, gamma, beta, xbf);
  cast_bf16_kernel<<<dim3((B_ * T_ * M_ * D_) / 1024), 256, 0, stream>>>(media, kvbf);
  frames_kernel<<<dim3(B_), 32, 0, stream>>>(loc, frame);

  const dim3 ggrid((B_ * N_) / 128, D_ / 64);
  gemm_wmma_kernel<<<ggrid, 256, 0, stream>>>(xbf,  wq, bq, qbf,  OUT_BF16);
  gemm_wmma_kernel<<<ggrid, 256, 0, stream>>>(kvbf, wk, bk, kbf,  OUT_BF16);
  gemm_wmma_kernel<<<ggrid, 256, 0, stream>>>(kvbf, wv, bv, vtbf, OUT_VT);

  attn_kernel<<<dim3(N_ / 16, H_, B_), 32, 0, stream>>>(qbf, kbf, vtbf, frame, obf);

  gemm_wmma_kernel<<<ggrid, 256, 0, stream>>>(obf, wout, bout, out, OUT_F32);
}
